// get_model_12824772345915
// MI455X (gfx1250) — matrix operand order repaired and checked by reading, not compiled, not run on hardware
//
#include <hip/hip_runtime.h>

typedef __attribute__((ext_vector_type(16))) _Float16 v16h;
typedef __attribute__((ext_vector_type(8)))  _Float16 h8;
typedef __attribute__((ext_vector_type(8)))  float    v8f;

#define BN_SCALE 0.9999950000374997f  // 1/sqrt(1+1e-5)

static __host__ __device__ inline int padk(int k){ return (k + 31) & ~31; }

// ---------------- transpose (B,3,N) -> (B,N,3), f32 ----------------
__global__ void k_transpose(const float* __restrict__ xyz, float* __restrict__ pts, int B, int N){
  int i = blockIdx.x*blockDim.x + threadIdx.x;
  if (i >= B*N) return;
  int b = i / N, n = i % N;
  const float* s = xyz + (size_t)b*3*N;
  float* d = pts + (size_t)i*3;
  d[0]=s[n]; d[1]=s[N+n]; d[2]=s[2*N+n];
}

// ---------------- farthest point sampling: one block per batch ----------------
__global__ void k_fps(const float* __restrict__ pts, float* __restrict__ centers, int N, int S){
  __shared__ float dist[2048];
  __shared__ float rv[256];
  __shared__ int   ri[256];
  __shared__ int   sel;
  int b = blockIdx.x, t = threadIdx.x, T = blockDim.x;
  const float* P = pts + (size_t)b*N*3;
  float* C = centers + (size_t)b*S*3;
  for (int i=t;i<N;i+=T) dist[i]=1e10f;
  if (t==0) sel = 0;
  __syncthreads();
  for (int it=0; it<S; ++it){
    int s = sel;
    float cx=P[s*3+0], cy=P[s*3+1], cz=P[s*3+2];
    if (t==0){ C[it*3+0]=cx; C[it*3+1]=cy; C[it*3+2]=cz; }
    float best=-1.0f; int bi=0x7fffffff;
    for (int i=t;i<N;i+=T){
      float dx=P[i*3+0]-cx, dy=P[i*3+1]-cy, dz=P[i*3+2]-cz;
      float d = dx*dx+dy*dy+dz*dz;
      float nd = fminf(dist[i], d);
      dist[i] = nd;
      if (nd > best){ best=nd; bi=i; }
    }
    rv[t]=best; ri[t]=bi;
    __syncthreads();
    for (int off=T>>1; off>0; off>>=1){
      if (t<off){
        if (rv[t+off] > rv[t] || (rv[t+off]==rv[t] && ri[t+off] < ri[t])){
          rv[t]=rv[t+off]; ri[t]=ri[t+off];
        }
      }
      __syncthreads();
    }
    if (t==0) sel = ri[0];
    __syncthreads();
  }
}

// ---------------- ball query (single batch): thread per center ----------------
__global__ void k_ball_query(const float* __restrict__ pts, const float* __restrict__ centers,
                             int* __restrict__ gidx, int N, int S, int K, float r2){
  int s = blockIdx.x*blockDim.x + threadIdx.x;
  if (s >= S) return;
  float cx=centers[s*3+0], cy=centers[s*3+1], cz=centers[s*3+2];
  int* g = gidx + (size_t)s*K;
  int cnt=0;
  for (int j=0;j<N && cnt<K;++j){
    float dx=pts[j*3+0]-cx, dy=pts[j*3+1]-cy, dz=pts[j*3+2]-cz;
    if (dx*dx+dy*dy+dz*dz <= r2) g[cnt++]=j;
  }
  int f = (cnt>0)? g[0] : 0;
  for (int k=cnt;k<K;++k) g[k]=f;
}

// ------- gather + concat([feats, xyz-center]) -> padded f16 rows (single batch) -------
__global__ void k_gather(const float* __restrict__ pts, const _Float16* __restrict__ feats,
                         const float* __restrict__ centers, const int* __restrict__ gidx,
                         _Float16* __restrict__ A, int S, int K, int Cf, int Kp){
  int row = blockIdx.x*blockDim.x + threadIdx.x;
  if (row >= S*K) return;
  int s = row / K;
  int g = gidx[row];
  _Float16* out = A + (size_t)row*Kp;
  for (int c=0;c<Cf;++c) out[c] = feats[(size_t)g*Cf + c];
  out[Cf+0]=(_Float16)(pts[g*3+0]-centers[s*3+0]);
  out[Cf+1]=(_Float16)(pts[g*3+1]-centers[s*3+1]);
  out[Cf+2]=(_Float16)(pts[g*3+2]-centers[s*3+2]);
  for (int c=Cf+3;c<Kp;++c) out[c]=(_Float16)0.0f;
}

// ---------------- weights: f32 (N,K) -> padded f16 (N,Kp) ----------------
__global__ void k_w2h(const float* __restrict__ W, _Float16* __restrict__ Wh,
                      int total, int K, int Kp){
  int i = blockIdx.x*blockDim.x + threadIdx.x;
  if (i >= total) return;
  int n = i / Kp, k = i % Kp;
  Wh[i] = (k < K) ? (_Float16)W[(size_t)n*K + k] : (_Float16)0.0f;
}

// ------- WMMA GEMM: out[M,cout] = act((A[M,Kin]*W[cout,Kin]^T + b)*scale) -------
// M mult of 16, cout mult of 16, Kin mult of 32 -> branch-free inner loop:
// 4x global_load_b128 + v_wmma_f32_16x16x32_f16 per K-step.
__global__ void k_gemm_wmma(const _Float16* __restrict__ A, const _Float16* __restrict__ W,
                            const float* __restrict__ bias, void* __restrict__ out,
                            int Kin, int cout, float scale, int relu, int out_f32){
  int m0 = blockIdx.x*16, n0 = blockIdx.y*16;
  int lane = threadIdx.x & 31;
  int hi = lane >> 4, r = lane & 15;
  const _Float16* Arow = A + (size_t)(m0 + r)*Kin + 8*hi;   // runs: [0..7], [16..23] (+8*hi)
  const _Float16* Wrow = W + (size_t)(n0 + r)*Kin + 8*hi;   // runs: [0..7], [16..23] (+8*hi)
  v8f acc = {};
  #pragma unroll 2
  for (int k0=0; k0<Kin; k0+=32){
    h8 a0 = *(const h8*)(Arow + k0);
    h8 a1 = *(const h8*)(Arow + k0 + 16);
    h8 b0 = *(const h8*)(Wrow + k0);
    h8 b1 = *(const h8*)(Wrow + k0 + 16);
    v16h af, bf;
    #pragma unroll
    for (int e=0;e<8;++e){ af[e]=a0[e]; af[8+e]=a1[e]; bf[e]=b0[e]; bf[8+e]=b1[e]; }
    acc = __builtin_amdgcn_wmma_f32_16x16x32_f16(
        /*neg_a=*/false, af, /*neg_b=*/false, bf,
        /*c_mod=*/(short)0, acc, /*reuse_a=*/false, /*reuse_b=*/false);
  }
  int ocol = n0 + r;
  float bb = bias[ocol];
  #pragma unroll
  for (int rr=0; rr<8; ++rr){
    int orow = m0 + rr + (hi ? 8 : 0);
    float v = (acc[rr] + bb) * scale;
    if (relu) v = fmaxf(v, 0.0f);
    if (out_f32) ((float*)out)[(size_t)orow*cout + ocol] = v;
    else ((_Float16*)out)[(size_t)orow*cout + ocol] = (_Float16)v;
  }
}

// ------- max over K: X f16 (R,K,C) -> optional f16 dst and/or f32 dst -------
__global__ void k_maxk(const _Float16* __restrict__ X, _Float16* __restrict__ Yh,
                       float* __restrict__ Yf, int R, int K, int C, int OC, int off){
  int i = blockIdx.x*blockDim.x + threadIdx.x;
  if (i >= R*C) return;
  int r = i / C, c = i % C;
  const _Float16* x = X + ((size_t)r*K)*C + c;
  float m = -3.402823466e38f;
  for (int k=0;k<K;++k) m = fmaxf(m, (float)x[(size_t)k*C]);
  if (Yh) Yh[(size_t)r*OC + off + c] = (_Float16)m;
  if (Yf) Yf[(size_t)r*OC + off + c] = m;
}

// ------- write l2_xyz into cols 0..2 of padded sa3 input, zero pad 643..671 -------
__global__ void k_a3_coords(const float* __restrict__ c2, _Float16* __restrict__ A3, int total){
  int i = blockIdx.x*blockDim.x + threadIdx.x;
  if (i >= total) return;
  _Float16* o = A3 + (size_t)i*672;
  o[0]=(_Float16)c2[i*3+0];
  o[1]=(_Float16)c2[i*3+1];
  o[2]=(_Float16)c2[i*3+2];
  for (int c=643;c<672;++c) o[c]=(_Float16)0.0f;
}

extern "C" void kernel_launch(void* const* d_in, const int* in_sizes, int n_in,
                              void* d_out, int out_size, void* d_ws, size_t ws_size,
                              hipStream_t stream){
  (void)in_sizes; (void)n_in; (void)out_size; (void)ws_size;
  const int B=16, N=2048;
  static const int sa1d[3][4] = {{3,32,32,64},{3,64,64,128},{3,64,96,128}};
  static const int sa2d[3][4] = {{323,64,64,128},{323,128,128,256},{323,128,128,256}};
  static const int sa3d[4]    = {643,256,512,1024};

  int p=0;
  const float* xyz = (const float*)d_in[p++];
  const float *sa1W[3][3], *sa1b[3][3], *sa2W[3][3], *sa2b[3][3], *sa3W[3], *sa3b[3];
  for (int br=0;br<3;++br) for(int l=0;l<3;++l){ sa1W[br][l]=(const float*)d_in[p++]; sa1b[br][l]=(const float*)d_in[p++]; }
  for (int br=0;br<3;++br) for(int l=0;l<3;++l){ sa2W[br][l]=(const float*)d_in[p++]; sa2b[br][l]=(const float*)d_in[p++]; }
  for (int l=0;l<3;++l){ sa3W[l]=(const float*)d_in[p++]; sa3b[l]=(const float*)d_in[p++]; }
  const float* fcW=(const float*)d_in[p++];
  const float* fcb=(const float*)d_in[p++];

  // ---- workspace carve-up (64B-aligned regions) ----
  char* cur = (char*)d_ws;
  auto carve = [&](size_t bytes)->void*{ void* q=(void*)cur; cur += (bytes + 63) & ~(size_t)63; return q; };
  float*    pts = (float*)   carve((size_t)B*N*3*sizeof(float));
  float*    c1  = (float*)   carve((size_t)B*300*3*sizeof(float));
  float*    c2  = (float*)   carve((size_t)B*128*3*sizeof(float));
  _Float16* l1p = (_Float16*)carve((size_t)B*300*320*sizeof(_Float16));
  _Float16* A3  = (_Float16*)carve((size_t)B*128*672*sizeof(_Float16));
  _Float16* bufA= (_Float16*)carve((size_t)6*1024*1024*sizeof(_Float16));
  _Float16* bufB= (_Float16*)carve((size_t)6*1024*1024*sizeof(_Float16));
  int*      gidx= (int*)     carve((size_t)300*128*sizeof(int));

  // ---- convert all weights to padded f16 ----
  auto wconv = [&](const float* W, int Nn, int K)->_Float16*{
    int Kp = padk(K);
    _Float16* Wh = (_Float16*)carve((size_t)Nn*Kp*sizeof(_Float16));
    int tot = Nn*Kp;
    k_w2h<<<(tot+255)/256,256,0,stream>>>(W, Wh, tot, K, Kp);
    return Wh;
  };
  _Float16 *sa1Wh[3][3], *sa2Wh[3][3], *sa3Wh[3], *fcWh;
  for (int br=0;br<3;++br) for(int l=0;l<3;++l) sa1Wh[br][l]=wconv(sa1W[br][l], sa1d[br][l+1], sa1d[br][l]);
  for (int br=0;br<3;++br) for(int l=0;l<3;++l) sa2Wh[br][l]=wconv(sa2W[br][l], sa2d[br][l+1], sa2d[br][l]);
  for (int l=0;l<3;++l) sa3Wh[l]=wconv(sa3W[l], sa3d[l+1], sa3d[l]);
  fcWh = wconv(fcW, 512, 1024);

  // ---- geometry ----
  { int tot=B*N; k_transpose<<<(tot+255)/256,256,0,stream>>>(xyz, pts, B, N); }
  k_fps<<<B,256,0,stream>>>(pts, c1, N, 300);

  // ---- SA1 (per batch) ----
  const float r1[3]={0.1f,0.2f,0.4f}; const int k1[3]={16,32,128};
  for (int b=0;b<B;++b){
    const float* Pb = pts + (size_t)b*N*3;
    const float* Cb = c1  + (size_t)b*300*3;
    _Float16* Ob = l1p + (size_t)b*300*320;
    int coff=0;
    for (int br=0;br<3;++br){
      int K=k1[br], S=300, rows=S*K;
      k_ball_query<<<(S+127)/128,128,0,stream>>>(Pb, Cb, gidx, N, S, K, r1[br]*r1[br]);
      int Kp0 = padk(sa1d[br][0]);
      k_gather<<<(rows+255)/256,256,0,stream>>>(Pb, nullptr, Cb, gidx, bufA, S, K, 0, Kp0);
      _Float16* in=bufA; _Float16* out=bufB;
      for (int l=0;l<3;++l){
        int Kin = (l==0)? Kp0 : sa1d[br][l];
        int cout = sa1d[br][l+1];
        dim3 g(rows/16, cout/16);
        k_gemm_wmma<<<g,32,0,stream>>>(in, sa1Wh[br][l], sa1b[br][l], out, Kin, cout, BN_SCALE, 1, 0);
        _Float16* t=in; in=out; out=t;
      }
      int cl=sa1d[br][3];
      k_maxk<<<(S*cl+255)/256,256,0,stream>>>(in, Ob, nullptr, S, K, cl, 320, coff);
      coff += cl;
    }
  }

  // ---- SA2 (per batch) ----
  k_fps<<<B,256,0,stream>>>(c1, c2, 300, 128);
  k_a3_coords<<<(B*128+255)/256,256,0,stream>>>(c2, A3, B*128);

  const float r2a[3]={0.2f,0.4f,0.8f}; const int k2[3]={32,64,128};
  for (int b=0;b<B;++b){
    const float* Pb = c1  + (size_t)b*300*3;
    const _Float16* Fb = l1p + (size_t)b*300*320;
    const float* Cb = c2  + (size_t)b*128*3;
    _Float16* Ob = A3 + (size_t)b*128*672;
    int coff=3;
    for (int br=0;br<3;++br){
      int K=k2[br], S=128, Npt=300, rows=S*K;
      k_ball_query<<<(S+127)/128,128,0,stream>>>(Pb, Cb, gidx, Npt, S, K, r2a[br]*r2a[br]);
      int Kp0 = padk(sa2d[br][0]);   // 352
      k_gather<<<(rows+255)/256,256,0,stream>>>(Pb, Fb, Cb, gidx, bufA, S, K, 320, Kp0);
      _Float16* in=bufA; _Float16* out=bufB;
      for (int l=0;l<3;++l){
        int Kin = (l==0)? Kp0 : sa2d[br][l];
        int cout = sa2d[br][l+1];
        dim3 g(rows/16, cout/16);
        k_gemm_wmma<<<g,32,0,stream>>>(in, sa2Wh[br][l], sa2b[br][l], out, Kin, cout, BN_SCALE, 1, 0);
        _Float16* t=in; in=out; out=t;
      }
      int cl=sa2d[br][3];
      k_maxk<<<(S*cl+255)/256,256,0,stream>>>(in, Ob, nullptr, S, K, cl, 672, coff);
      coff += cl;
    }
  }

  // ---- SA3 (all batches) + global max + fc head ----
  {
    int rows=B*128;
    const _Float16* in = A3;
    _Float16* pp[2] = {bufA, bufB};
    for (int l=0;l<3;++l){
      int Kin = (l==0)? 672 : sa3d[l];
      int cout = sa3d[l+1];
      _Float16* out = pp[l&1];                 // l0->bufA, l1->bufB, l2->bufA
      dim3 g(rows/16, cout/16);
      k_gemm_wmma<<<g,32,0,stream>>>(in, sa3Wh[l], sa3b[l], out, Kin, cout, BN_SCALE, 1, 0);
      in = out;
    }
    // in == bufA (rows=2048, C=1024). Max over the 128 points per batch.
    float* l3f = (float*)d_out + (size_t)B*512;   // second output region (16,1024)
    k_maxk<<<(B*1024+255)/256,256,0,stream>>>(bufA, bufB, l3f, B, 128, 1024, 1024, 0);
    // fc: (16,1024) @ (512,1024)^T + b -> f32 d_out
    dim3 g(B/16, 512/16);
    k_gemm_wmma<<<g,32,0,stream>>>(bufB, fcWh, fcb, d_out, 1024, 512, 1.0f, 0, 1);
  }
}
